// MambaTS_85005992722696
// MI455X (gfx1250) — compile-verified, matrix operand order checked by reading, not run on hardware
//
#include <hip/hip_runtime.h>
#include <hip/hip_bf16.h>
#include <math.h>

// ---------------- model constants ----------------
#define Bc     4
#define Tc     1024
#define DIN    32
#define DM     768
#define Ic     1536
#define TWO_I  3072
#define DST    16
#define DTR    48
#define SPD    80          // DT_RANK + 2*D_STATE
#define KCONV  4
#define MROWS  (Bc*Tc)     // 4096
#define EPSV   1e-5f

typedef __attribute__((ext_vector_type(16))) _Float16 v16h;
typedef __attribute__((ext_vector_type(8)))  _Float16 v8h;
typedef __attribute__((ext_vector_type(8)))  float    v8f;

// ---------------- f32 -> f16 convert ----------------
__global__ void cvt_f16_kernel(const float* __restrict__ in, _Float16* __restrict__ out, int n) {
    int i = blockIdx.x * blockDim.x + threadIdx.x;
    int stride = gridDim.x * blockDim.x;
    for (; i < n; i += stride) out[i] = (_Float16)in[i];
}

// ---------------- WMMA GEMM:  C[M,N] = A[M,K] * W[N,K]^T (+bias)(+Cin) ----------------
// A, W: f16 row-major, K-contiguous, K % 32 == 0, M % 16 == 0, N % 16 == 0.
// One wave per 16x16 tile; 8 waves per block.
__global__ __launch_bounds__(256)
void wmma_gemm_nt(const _Float16* __restrict__ A, const _Float16* __restrict__ W,
                  const float* __restrict__ bias, const float* __restrict__ Cin,
                  float* __restrict__ Cout, int M, int N, int Kd) {
    const int lane  = threadIdx.x & 31;
    const int wave  = threadIdx.x >> 5;
    const int tilesM = M >> 4, tilesN = N >> 4;
    const int tile = blockIdx.x * 8 + wave;
    if (tile >= tilesM * tilesN) return;
    const int tm = tile % tilesM;
    const int tn = tile / tilesM;

    const int lrow  = lane & 15;      // m (for A) / n (for B) within tile
    const int lhalf = lane >> 4;      // selects K sub-group per ISA layout

    const _Float16* Ap = A + (size_t)(tm * 16 + lrow) * Kd + lhalf * 8;
    const _Float16* Wp = W + (size_t)(tn * 16 + lrow) * Kd + lhalf * 8;

    const int colg = tn * 16 + lrow;
    v8f c;
    #pragma unroll
    for (int r = 0; r < 8; ++r) {
        const int mg = tm * 16 + r + lhalf * 8;
        float v = 0.0f;
        if (Cin)  v += Cin[(size_t)mg * N + colg];
        if (bias) v += bias[colg];
        c[r] = v;
    }

    for (int k0 = 0; k0 < Kd; k0 += 32) {
        v8h a0 = *(const v8h*)(Ap + k0);
        v8h a1 = *(const v8h*)(Ap + k0 + 16);
        v8h b0 = *(const v8h*)(Wp + k0);
        v8h b1 = *(const v8h*)(Wp + k0 + 16);
        v16h av, bv;
        #pragma unroll
        for (int j = 0; j < 8; ++j) { av[j] = a0[j]; av[j + 8] = a1[j]; }
        #pragma unroll
        for (int j = 0; j < 8; ++j) { bv[j] = b0[j]; bv[j + 8] = b1[j]; }
        c = __builtin_amdgcn_wmma_f32_16x16x32_f16(false, av, false, bv,
                                                   (short)0, c, false, false);
    }

    #pragma unroll
    for (int r = 0; r < 8; ++r) {
        const int mg = tm * 16 + r + lhalf * 8;
        Cout[(size_t)mg * N + colg] = c[r];
    }
}

// ---------------- rmsnorm -> f16 (one block per row) ----------------
__global__ __launch_bounds__(256)
void rmsnorm_f16_kernel(const float* __restrict__ h, const float* __restrict__ w,
                        _Float16* __restrict__ out) {
    __shared__ float red[256];
    const int row = blockIdx.x, tid = threadIdx.x;
    const float* hr = h + (size_t)row * DM;
    float ss = 0.0f;
    for (int c = tid; c < DM; c += 256) { float v = hr[c]; ss += v * v; }
    red[tid] = ss; __syncthreads();
    for (int s = 128; s > 0; s >>= 1) { if (tid < s) red[tid] += red[tid + s]; __syncthreads(); }
    const float scale = rsqrtf(red[0] / (float)DM + EPSV);
    for (int c = tid; c < DM; c += 256)
        out[(size_t)row * DM + c] = (_Float16)(hr[c] * scale * w[c]);
}

// ---------------- causal depthwise conv (K=4) + silu ----------------
__global__ __launch_bounds__(256)
void conv_silu_kernel(const float* __restrict__ proj, const float* __restrict__ cw,
                      const float* __restrict__ cb, float* __restrict__ u,
                      _Float16* __restrict__ u16) {
    const int gid = blockIdx.x * 256 + threadIdx.x;
    if (gid >= MROWS * Ic) return;
    const int i  = gid % Ic;
    const int bt = gid / Ic;
    const int t  = bt % Tc;
    const int b  = bt / Tc;
    float acc = cb[i];
    #pragma unroll
    for (int k = 0; k < KCONV; ++k) {
        const int tp = t - (KCONV - 1) + k;
        if (tp >= 0)
            acc += cw[i * KCONV + k] * proj[((size_t)(b * Tc + tp)) * TWO_I + i];
    }
    const float v = acc / (1.0f + expf(-acc));  // silu
    u[gid] = v;
    u16[gid] = (_Float16)v;
}

// ---------------- dt = softplus(dtr @ dt_w^T + dt_b) (one block per (b,t)) ----------------
__global__ __launch_bounds__(256)
void dt_softplus_kernel(const float* __restrict__ sp, const float* __restrict__ dtw,
                        const float* __restrict__ dtb, float* __restrict__ dt) {
    __shared__ float sdt[DTR];
    const int row = blockIdx.x, tid = threadIdx.x;
    if (tid < DTR) sdt[tid] = sp[(size_t)row * SPD + tid];
    __syncthreads();
    for (int i = tid; i < Ic; i += 256) {
        float acc = dtb[i];
        #pragma unroll 8
        for (int r = 0; r < DTR; ++r) acc += sdt[r] * dtw[i * DTR + r];
        dt[(size_t)row * Ic + i] = (acc > 20.0f) ? acc : log1pf(expf(acc));
    }
}

// ---------------- selective scan (sequential over T) ----------------
// grid: (Ic/256, Bc); states in registers; B/C vectors staged in LDS per step.
__global__ __launch_bounds__(256)
void scan_kernel(const float* __restrict__ dtg, const float* __restrict__ u,
                 const float* __restrict__ sp, const float* __restrict__ proj,
                 const float* __restrict__ alog, const float* __restrict__ dp,
                 _Float16* __restrict__ y16) {
    __shared__ float shm[2 * DST];   // [0:16] B_t, [16:32] C_t
    const int tid = threadIdx.x;
    const int i = blockIdx.x * 256 + tid;
    const int b = blockIdx.y;
    float Aa[DST], s[DST];
    #pragma unroll
    for (int n = 0; n < DST; ++n) { Aa[n] = -expf(alog[(size_t)i * DST + n]); s[n] = 0.0f; }
    const float dpv = dp[i];
    for (int t = 0; t < Tc; ++t) {
        const size_t rbt = (size_t)(b * Tc + t);
        if (tid < 2 * DST) shm[tid] = sp[rbt * SPD + DTR + tid];
        __syncthreads();
        const size_t idx = rbt * Ic + i;
        const float dtv = dtg[idx];
        const float uv  = u[idx];
        const float du  = dtv * uv;
        float y = 0.0f;
        #pragma unroll
        for (int n = 0; n < DST; ++n) {
            s[n] = s[n] * expf(dtv * Aa[n]) + du * shm[n];
            y += s[n] * shm[DST + n];
        }
        y += uv * dpv;
        const float g = proj[rbt * TWO_I + Ic + i];
        y *= g / (1.0f + expf(-g));   // * silu(gate)
        y16[idx] = (_Float16)y;
        __syncthreads();
    }
}

// ---------------- final rmsnorm + out projection (N=1) ----------------
__global__ __launch_bounds__(256)
void final_out_kernel(const float* __restrict__ h, const float* __restrict__ nw,
                      const float* __restrict__ Wout, const float* __restrict__ bout,
                      float* __restrict__ out) {
    __shared__ float red[256];
    const int row = blockIdx.x, tid = threadIdx.x;
    const float* hr = h + (size_t)row * DM;
    float ss = 0.0f;
    for (int c = tid; c < DM; c += 256) { float v = hr[c]; ss += v * v; }
    red[tid] = ss; __syncthreads();
    for (int s = 128; s > 0; s >>= 1) { if (tid < s) red[tid] += red[tid + s]; __syncthreads(); }
    const float scale = rsqrtf(red[0] / (float)DM + EPSV);
    __syncthreads();
    float dot = 0.0f;
    for (int c = tid; c < DM; c += 256) dot += hr[c] * nw[c] * Wout[c];
    red[tid] = dot; __syncthreads();
    for (int s = 128; s > 0; s >>= 1) { if (tid < s) red[tid] += red[tid + s]; __syncthreads(); }
    if (tid == 0) out[row] = red[0] * scale + bout[0];
}

// ---------------- host launch ----------------
extern "C" void kernel_launch(void* const* d_in, const int* in_sizes, int n_in,
                              void* d_out, int out_size, void* d_ws, size_t ws_size,
                              hipStream_t stream) {
    (void)in_sizes; (void)n_in; (void)out_size; (void)ws_size;
    const float* x        = (const float*)d_in[0];
    const float* W_in     = (const float*)d_in[1];
    const float* b_in     = (const float*)d_in[2];
    const float* norm_w   = (const float*)d_in[3];
    const float* mix_in_w = (const float*)d_in[4];
    const float* conv_w   = (const float*)d_in[5];
    const float* conv_b   = (const float*)d_in[6];
    const float* xproj_w  = (const float*)d_in[7];
    const float* dt_w     = (const float*)d_in[8];
    const float* dt_b     = (const float*)d_in[9];
    const float* A_log    = (const float*)d_in[10];
    const float* D_p      = (const float*)d_in[11];
    const float* mix_out_w= (const float*)d_in[12];
    const float* normf_w  = (const float*)d_in[13];
    const float* W_out    = (const float*)d_in[14];
    const float* b_out    = (const float*)d_in[15];
    float* out = (float*)d_out;

    // workspace carve-up (256B aligned)
    char* base = (char*)d_ws;
    size_t off = 0;
    auto alloc = [&](size_t bytes) -> void* {
        void* p = base + off;
        off += (bytes + 255) & ~(size_t)255;
        return p;
    };
    _Float16* x16    = (_Float16*)alloc((size_t)MROWS * DIN * 2);
    _Float16* Win16  = (_Float16*)alloc((size_t)DM * DIN * 2);
    _Float16* min16  = (_Float16*)alloc((size_t)2 * TWO_I * DM * 2);
    _Float16* xp16   = (_Float16*)alloc((size_t)2 * SPD * Ic * 2);
    _Float16* mo16   = (_Float16*)alloc((size_t)2 * DM * Ic * 2);
    float*    h      = (float*)   alloc((size_t)MROWS * DM * 4);
    _Float16* hn16   = (_Float16*)alloc((size_t)MROWS * DM * 2);
    float*    proj   = (float*)   alloc((size_t)MROWS * TWO_I * 4);
    float*    ub     = (float*)   alloc((size_t)MROWS * Ic * 4);
    _Float16* u16    = (_Float16*)alloc((size_t)MROWS * Ic * 2);
    float*    spb    = (float*)   alloc((size_t)MROWS * SPD * 4);
    float*    dtb_f  = (float*)   alloc((size_t)MROWS * Ic * 4);
    _Float16* y16    = (_Float16*)alloc((size_t)MROWS * Ic * 2);

    auto cvt = [&](const float* src, _Float16* dst, int n) {
        int blocks = (n + 255) / 256; if (blocks > 4096) blocks = 4096;
        hipLaunchKernelGGL(cvt_f16_kernel, dim3(blocks), dim3(256), 0, stream, src, dst, n);
    };
    cvt(x,         x16,   MROWS * DIN);
    cvt(W_in,      Win16, DM * DIN);
    cvt(mix_in_w,  min16, 2 * TWO_I * DM);
    cvt(xproj_w,   xp16,  2 * SPD * Ic);
    cvt(mix_out_w, mo16,  2 * DM * Ic);

    auto gemm = [&](const _Float16* A, const _Float16* W, const float* bias,
                    const float* Cin, float* Cout, int M, int N, int K) {
        int tiles = (M / 16) * (N / 16);
        hipLaunchKernelGGL(wmma_gemm_nt, dim3((tiles + 7) / 8), dim3(256), 0, stream,
                           A, W, bias, Cin, Cout, M, N, K);
    };

    // h = x @ W_in^T + b_in
    gemm(x16, Win16, b_in, nullptr, h, MROWS, DM, DIN);

    for (int l = 0; l < 2; ++l) {
        // hn16 = rmsnorm(h, norm_w[l]) as f16
        hipLaunchKernelGGL(rmsnorm_f16_kernel, dim3(MROWS), dim3(256), 0, stream,
                           h, norm_w + (size_t)l * DM, hn16);
        // proj = hn @ mix_in_w[l]^T  (4096 x 3072, K=768)
        gemm(hn16, min16 + (size_t)l * TWO_I * DM, nullptr, nullptr, proj, MROWS, TWO_I, DM);
        // u = silu(causal depthwise conv(proj[:, :I]) + conv_b)
        hipLaunchKernelGGL(conv_silu_kernel, dim3((MROWS * Ic + 255) / 256), dim3(256), 0, stream,
                           proj, conv_w + (size_t)l * Ic * KCONV, conv_b + (size_t)l * Ic, ub, u16);
        // sp = u @ xproj_w[l]^T  (4096 x 80, K=1536)
        gemm(u16, xp16 + (size_t)l * SPD * Ic, nullptr, nullptr, spb, MROWS, SPD, Ic);
        // dt = softplus(sp[:, :48] @ dt_w[l]^T + dt_b[l])
        hipLaunchKernelGGL(dt_softplus_kernel, dim3(MROWS), dim3(256), 0, stream,
                           spb, dt_w + (size_t)l * Ic * DTR, dt_b + (size_t)l * Ic, dtb_f);
        // selective scan -> y16 (includes +u*D, *silu(gate))
        hipLaunchKernelGGL(scan_kernel, dim3(Ic / 256, Bc), dim3(256), 0, stream,
                           dtb_f, ub, spb, proj,
                           A_log + (size_t)l * Ic * DST, D_p + (size_t)l * Ic, y16);
        // h += y @ mix_out_w[l]^T  (residual via Cin=h)
        gemm(y16, mo16 + (size_t)l * DM * Ic, nullptr, h, h, MROWS, DM, Ic);
    }

    // out = rmsnorm(h, normf_w) @ W_out^T + b_out
    hipLaunchKernelGGL(final_out_kernel, dim3(MROWS), dim3(256), 0, stream,
                       h, normf_w, W_out, b_out, out);
}
